// ShuffleSelfAttentionLayerV2_84310208020958
// MI455X (gfx1250) — matrix operand order repaired and checked by reading, not compiled, not run on hardware
//
#include <hip/hip_runtime.h>
#include <cstdint>
#include <cstddef>

typedef __attribute__((ext_vector_type(16))) _Float16 v16h;
typedef __attribute__((ext_vector_type(8)))  float    v8f;
typedef __attribute__((ext_vector_type(4)))  float    v4f;
typedef __attribute__((ext_vector_type(2)))  float    v2f;

#define HB_   64      // H*B batched attention problems
#define LTOT  1024    // sequence length per problem
#define DDIM  128     // head dim
#define NN    64
#define FF    512
#define BB    8
#define CTC   3
#define INV_TEMP 0.125f

// -------------------------------------------------------------------------
// Kernel 1: gather + 3-channel linear, packed directly into WMMA fragment
// layout (ISA 7.12.2).  One thread produces one lane's v16h (32B store).
//   Q: A-frag 16x32 f16   [hb][ltile(64)][kc(4)][lane(32)][16h]
//   K: B-frag 32x16 f16   [hb][ct(64)][kc(4)][lane(32)][16h]   (K^T columns)
//   V: B-frag 32x16 f16   [hb][mc(32)][dtile(8)][lane(32)][16h]
// -------------------------------------------------------------------------
__global__ void qkv_pack_kernel(const float* __restrict__ feat,
                                const int*   __restrict__ index,
                                const float* __restrict__ Wq,
                                const float* __restrict__ Wk,
                                const float* __restrict__ Wv,
                                _Float16* __restrict__ Qf,
                                _Float16* __restrict__ Kf,
                                _Float16* __restrict__ Vf) {
  const int tid     = blockIdx.x * blockDim.x + threadIdx.x;   // 3 * 524288
  const int sel     = tid >> 19;                // 0=Q 1=K 2=V (uniform per block)
  const int r       = tid & ((1 << 19) - 1);
  const int lane    = r & 31;
  const int fragIdx = r >> 5;
  const int laneM   = lane & 15;
  const int lanehi  = lane >> 4;
  const int hb      = fragIdx >> 8;             // 256 frags per hb in all layouts
  const int h = hb >> 3, b = hb & 7;

  const float* W    = (sel == 0) ? Wq : (sel == 1) ? Wk : Wv;
  _Float16*    dst  = (sel == 0) ? Qf : (sel == 1) ? Kf : Vf;

  v16h out;
#pragma unroll
  for (int p = 0; p < 16; ++p) {
    int l, d;
    if (sel == 0) {               // A-fragment (rows = laneM)
      const int kc = fragIdx & 3, ltile = (fragIdx >> 2) & 63;
      const int i = p >> 1, j = p & 1;
      const int k = ((i >= 4) ? 16 : 0) + (i & 3) * 2 + lanehi * 8 + j;
      l = ltile * 16 + laneM;  d = kc * 32 + k;
    } else if (sel == 1) {        // B-fragment of K^T (cols = laneM)
      const int kc = fragIdx & 3, ct = (fragIdx >> 2) & 63;
      const int v = p >> 1, j = p & 1;
      const int k = ((v >= 4) ? 16 : 0) + (v & 3) * 2 + lanehi * 8 + j;
      l = ct * 16 + laneM;  d = kc * 32 + k;
    } else {                      // B-fragment of V (cols = d)
      const int dtile = fragIdx & 7, mc = (fragIdx >> 3) & 31;
      const int v = p >> 1, j = p & 1;
      l = mc * 32 + ((v >= 4) ? 16 : 0) + (v & 3) * 2 + lanehi * 8 + j;
      d = dtile * 16 + laneM;
    }
    // blockified element (hb, l, d) -> (b, n, o=h*4+cc, f)
    const int n  = l >> 4,  fb = l & 15;
    const int fr = d >> 2,  cc = d & 3;
    const int f  = fb * 32 + fr;
    const int idx = index[n * FF + f];
    const int o  = h * 4 + cc;
    const float s = feat[(b * CTC + 0) * FF + idx] * W[o * CTC + 0]
                  + feat[(b * CTC + 1) * FF + idx] * W[o * CTC + 1]
                  + feat[(b * CTC + 2) * FF + idx] * W[o * CTC + 2];
    out[p] = (_Float16)s;
  }
  *(v16h*)(dst + (size_t)r * 16) = out;
}

// -------------------------------------------------------------------------
// Kernel 2: attention.  Block = 2 waves; each wave owns 16 rows and a
// 16x1024 f32 score stripe in LDS (64KB/wave) + 32-float row-stat array.
// All LDS traffic is wave-private -> no barriers needed (DS ops in-order).
// -------------------------------------------------------------------------
__global__ void attn_kernel(const _Float16* __restrict__ Qfr,
                            const _Float16* __restrict__ Kfr,
                            const _Float16* __restrict__ Vfr,
                            float* __restrict__ attn,
                            float* __restrict__ Obuf) {
  extern __shared__ float lds[];
  const int hb     = blockIdx.y;
  const int wave   = threadIdx.x >> 5;
  const int lane   = threadIdx.x & 31;
  const int laneM  = lane & 15, lanehi = lane >> 4;
  float* S      = lds + wave * (16 * LTOT);
  float* rowred = lds + 2 * (16 * LTOT) + wave * 32;  // [0:16) rowmax, [16:32) rinv
  const int rowtile = blockIdx.x * 2 + wave;   // 0..63 (16-row tile index)

  const v16h* Qv = (const v16h*)Qfr;
  const v16h* Kv = (const v16h*)Kfr;
  const v16h* Vv = (const v16h*)Vfr;

  // Q fragments for all 4 K-chunks (K dim 128 = 4 x 32)
  v16h qf[4];
#pragma unroll
  for (int kc = 0; kc < 4; ++kc)
    qf[kc] = Qv[(size_t)(((hb * 64 + rowtile) * 4) + kc) * 32 + lane];

  float rmax[8];
#pragma unroll
  for (int r2 = 0; r2 < 8; ++r2) rmax[r2] = -3.0e38f;

  // ---- Phase A: S = (Q K^T)/temp into LDS, track running row max --------
  for (int ct = 0; ct < 64; ++ct) {
    v8f acc = {};
    const size_t kb = (size_t)((hb * 64 + ct) * 4) * 32 + lane;
#pragma unroll
    for (int kc = 0; kc < 4; ++kc)
      acc = __builtin_amdgcn_wmma_f32_16x16x32_f16(
          false, qf[kc], false, Kv[kb + kc * 32], (short)0, acc, false, false);
    if (ct < 63)
      __builtin_prefetch(&Kv[kb + 4 * 32], 0, 1);   // global_prefetch_b8
#pragma unroll
    for (int r2 = 0; r2 < 8; ++r2) {                // D layout: M=r2+8*lanehi, N=laneM
      const float s = acc[r2] * INV_TEMP;
      rmax[r2] = fmaxf(rmax[r2], s);
      S[(lanehi * 8 + r2) * LTOT + ct * 16 + laneM] = s;
    }
  }
  // row max across the 16 lanes of each half-wave, park in LDS per row
#pragma unroll
  for (int r2 = 0; r2 < 8; ++r2) {
    float m = rmax[r2];
    m = fmaxf(m, __shfl_xor(m, 1));
    m = fmaxf(m, __shfl_xor(m, 2));
    m = fmaxf(m, __shfl_xor(m, 4));
    m = fmaxf(m, __shfl_xor(m, 8));
    rmax[r2] = m;
  }
  if (laneM == 0) {
#pragma unroll
    for (int r2 = 0; r2 < 8; ++r2) rowred[lanehi * 8 + r2] = rmax[r2];
  }

  // ---- Phase B: per-row coalesced exp + row-sum (b128 LDS traffic) ------
  for (int row = 0; row < 16; ++row) {
    const float m = rowred[row];
    float* srow = S + row * LTOT;
    float psum = 0.f;
#pragma unroll
    for (int step = 0; step < 8; ++step) {
      v4f v = *(const v4f*)(srow + step * 128 + lane * 4);
      v4f e;
#pragma unroll
      for (int c = 0; c < 4; ++c) e[c] = __expf(v[c] - m);
      psum += e[0] + e[1] + e[2] + e[3];
      *(v4f*)(srow + step * 128 + lane * 4) = e;
    }
    psum += __shfl_xor(psum, 1);
    psum += __shfl_xor(psum, 2);
    psum += __shfl_xor(psum, 4);
    psum += __shfl_xor(psum, 8);
    psum += __shfl_xor(psum, 16);
    if (lane == 0) rowred[16 + row] = 1.0f / psum;
  }

  // ---- Phase C: normalized attention, coalesced b128 stores (256MB) -----
  for (int row = 0; row < 16; ++row) {
    const float inv = rowred[16 + row];
    const float* srow = S + row * LTOT;
    float* drow = attn + ((size_t)hb * LTOT + rowtile * 16 + row) * LTOT;
#pragma unroll
    for (int step = 0; step < 8; ++step) {
      v4f v = *(const v4f*)(srow + step * 128 + lane * 4);
      v4f o; o[0] = v[0] * inv; o[1] = v[1] * inv; o[2] = v[2] * inv; o[3] = v[3] * inv;
      *(v4f*)(drow + step * 128 + lane * 4) = o;
    }
  }

  // ---- Phase D: O = softmax(S) @ V  (fold 1/rowsum into final scale) ----
  v8f oacc[8] = {};
  for (int mc = 0; mc < 32; ++mc) {
    v16h pa;                        // A-fragment of P (16 x 32) from LDS
#pragma unroll
    for (int i = 0; i < 8; ++i) {   // adjacent column pairs -> ds_load_b64
      const int kbase = ((i >= 4) ? 16 : 0) + (i & 3) * 2 + lanehi * 8;
      v2f pv = *(const v2f*)(S + laneM * LTOT + mc * 32 + kbase);
      pa[2 * i]     = (_Float16)pv[0];
      pa[2 * i + 1] = (_Float16)pv[1];
    }
    const size_t vb = (size_t)((hb * 32 + mc) * 8) * 32 + lane;
#pragma unroll
    for (int dt = 0; dt < 8; ++dt)
      oacc[dt] = __builtin_amdgcn_wmma_f32_16x16x32_f16(
          false, pa, false, Vv[vb + dt * 32], (short)0, oacc[dt], false, false);
  }
  // scale by 1/rowsum, stage through LDS (scores dead now), store coalesced
  float rinv2[8];
#pragma unroll
  for (int r2 = 0; r2 < 8; ++r2) rinv2[r2] = rowred[16 + lanehi * 8 + r2];
#pragma unroll
  for (int dt = 0; dt < 8; ++dt) {
#pragma unroll
    for (int r2 = 0; r2 < 8; ++r2)
      S[(lanehi * 8 + r2) * DDIM + dt * 16 + laneM] = oacc[dt][r2] * rinv2[r2];
  }
  for (int row = 0; row < 16; ++row) {
    v4f v = *(const v4f*)(S + row * DDIM + lane * 4);
    *(v4f*)(Obuf + ((size_t)hb * LTOT + rowtile * 16 + row) * DDIM + lane * 4) = v;
  }
}

// -------------------------------------------------------------------------
// Kernel 3: channel-shuffle + conv2d reduction over k = h*4 + (fb>>2)
//   out3[b][n][f] = sum_h sum_q w2[h*4+q] * O[h*8+b][n*16 + q*4 + f>>7][f&127]
// -------------------------------------------------------------------------
__global__ void conv2d_kernel(const float* __restrict__ Obuf,
                              const float* __restrict__ w2,
                              float* __restrict__ out3) {
  const int t = blockIdx.x * blockDim.x + threadIdx.x;
  if (t >= BB * NN * FF) return;
  const int f = t & 511, n = (t >> 9) & 63, b = t >> 15;
  const int g = f >> 7, e = f & 127;
  float s = 0.f;
#pragma unroll
  for (int h = 0; h < 8; ++h)
#pragma unroll
    for (int q = 0; q < 4; ++q)
      s += w2[h * 4 + q] *
           Obuf[((size_t)(h * 8 + b) * LTOT + n * 16 + q * 4 + g) * DDIM + e];
  out3[t] = s;
}

// -------------------------------------------------------------------------
// Kernel 4: conv1d / mean / max pooling over n, + residual
// -------------------------------------------------------------------------
__global__ void pool_kernel(const float* __restrict__ out3,
                            const float* __restrict__ w1,
                            const float* __restrict__ feat,
                            float* __restrict__ ypre) {
  const int t = blockIdx.x * blockDim.x + threadIdx.x;   // 8*512
  if (t >= BB * FF) return;
  const int f = t & 511, b = t >> 9;
  float conv = 0.f, avg = 0.f, mx = -3.0e38f;
  for (int n = 0; n < NN; ++n) {
    const float v = out3[((size_t)b * NN + n) * FF + f];
    conv += v * w1[n];
    avg  += v;
    mx    = fmaxf(mx, v);
  }
  ypre[(b * 3 + 0) * FF + f] = conv             + feat[(b * 3 + 0) * FF + f];
  ypre[(b * 3 + 1) * FF + f] = avg * (1.f / NN) + feat[(b * 3 + 1) * FF + f];
  ypre[(b * 3 + 2) * FF + f] = mx               + feat[(b * 3 + 2) * FF + f];
}

// -------------------------------------------------------------------------
// Kernel 5: LayerNorm over F per (b, ct) row
// -------------------------------------------------------------------------
__global__ void ln_kernel(const float* __restrict__ ypre,
                          const float* __restrict__ gamma,
                          const float* __restrict__ beta,
                          float* __restrict__ yout) {
  __shared__ float red0[256];
  __shared__ float red1[256];
  const int row = blockIdx.x;       // 0..23
  const int t   = threadIdx.x;      // 256
  const float x0 = ypre[row * FF + t];
  const float x1 = ypre[row * FF + t + 256];
  red0[t] = x0 + x1;
  red1[t] = x0 * x0 + x1 * x1;
  __syncthreads();
  for (int off = 128; off > 0; off >>= 1) {
    if (t < off) { red0[t] += red0[t + off]; red1[t] += red1[t + off]; }
    __syncthreads();
  }
  const float mu   = red0[0] * (1.f / FF);
  const float var  = red1[0] * (1.f / FF) - mu * mu;
  const float rstd = rsqrtf(var + 1e-5f);
  yout[row * FF + t]       = (x0 - mu) * rstd * gamma[t]       + beta[t];
  yout[row * FF + t + 256] = (x1 - mu) * rstd * gamma[t + 256] + beta[t + 256];
}

// -------------------------------------------------------------------------
extern "C" void kernel_launch(void* const* d_in, const int* in_sizes, int n_in,
                              void* d_out, int out_size, void* d_ws, size_t ws_size,
                              hipStream_t stream) {
  const float* feat  = (const float*)d_in[0];
  const int*   index = (const int*)  d_in[1];
  const float* Wq    = (const float*)d_in[2];
  const float* Wk    = (const float*)d_in[3];
  const float* Wv    = (const float*)d_in[4];
  const float* w2    = (const float*)d_in[5];
  const float* w1    = (const float*)d_in[6];
  const float* gamma = (const float*)d_in[7];
  const float* beta  = (const float*)d_in[8];

  char* ws = (char*)d_ws;
  _Float16* Qf   = (_Float16*)(ws + (size_t)0);
  _Float16* Kf   = (_Float16*)(ws + (size_t)16 * 1024 * 1024);
  _Float16* Vf   = (_Float16*)(ws + (size_t)32 * 1024 * 1024);
  float*    Obuf = (float*)   (ws + (size_t)48 * 1024 * 1024);   // 32MB
  float*    out3 = (float*)   (ws + (size_t)80 * 1024 * 1024);   // 1MB
  float*    ypre = (float*)   (ws + (size_t)81 * 1024 * 1024);   // 48KB

  float* yout = (float*)d_out;                          // (8,3,512)
  float* attn = (float*)d_out + (size_t)BB * CTC * FF;  // (64,1024,1024)

  qkv_pack_kernel<<<(3 * 524288) / 256, 256, 0, stream>>>(
      feat, index, Wq, Wk, Wv, Qf, Kf, Vf);

  dim3 grid2(32, HB_);
  const size_t ldsBytes = (2 * 16 * LTOT + 64) * sizeof(float);  // 131328
  attn_kernel<<<grid2, 64, ldsBytes, stream>>>(Qf, Kf, Vf, attn, Obuf);

  conv2d_kernel<<<(BB * NN * FF + 255) / 256, 256, 0, stream>>>(Obuf, w2, out3);
  pool_kernel<<<(BB * FF + 255) / 256, 256, 0, stream>>>(out3, w1, feat, ypre);
  ln_kernel<<<BB * CTC, 256, 0, stream>>>(ypre, gamma, beta, yout);
}
